// GraphTransformerLayer_73143293051182
// MI455X (gfx1250) — matrix operand order repaired and checked by reading, not compiled, not run on hardware
//
#include <hip/hip_runtime.h>
#include <hip/hip_bf16.h>

typedef __attribute__((ext_vector_type(16))) __bf16 v16bf;
typedef __attribute__((ext_vector_type(8)))  __bf16 v8bf;
typedef __attribute__((ext_vector_type(8)))  float  v8f;

#define B_  4
#define N_  2000
#define K_  30
#define H_  128
#define NH_ 4
#define NIN_ 256
#define DFF_ 512
#define EPS_ 1e-6f

#define HE_LD 264   // bf16 elems per padded h_e row (528 B, 16B-aligned rows)
#define KV_LD 132   // f32 elems per padded K'/V' row

// ---------------- weight conversion: f32 -> bf16 ----------------
__global__ void cvt_weights(const float* __restrict__ wk, const float* __restrict__ wv,
                            __bf16* __restrict__ wkb, __bf16* __restrict__ wvb) {
    int i = blockIdx.x * blockDim.x + threadIdx.x;
    if (i < H_ * NIN_) {
        wkb[i] = (__bf16)wk[i];
        wvb[i] = (__bf16)wv[i];
    }
}

// ---------------- fused graph-transformer layer, one WG per (b,n) ----------------
__global__ __launch_bounds__(256)
void gt_layer(const float* __restrict__ hv, const float* __restrict__ he,
              const float* __restrict__ mask_v, const float* __restrict__ mask_attend,
              const float* __restrict__ WQ, const float* __restrict__ WO,
              const float* __restrict__ gain0, const float* __restrict__ bias0,
              const float* __restrict__ gain1, const float* __restrict__ bias1,
              const float* __restrict__ Win, const float* __restrict__ bin,
              const float* __restrict__ Wout, const float* __restrict__ bout,
              const __bf16* __restrict__ WKb, const __bf16* __restrict__ WVb,
              float* __restrict__ out) {
    __shared__ __attribute__((aligned(32))) __bf16 sHe[32 * HE_LD];
    __shared__ float sK[32 * KV_LD];
    __shared__ float sV[32 * KV_LD];
    __shared__ float sHv[H_];
    __shared__ float sQ[H_];
    __shared__ float sMask[32];
    __shared__ float sLog[NH_ * 32];
    __shared__ float sAtt[NH_ * 32];
    __shared__ float sHupd[H_];
    __shared__ float sH[H_];
    __shared__ float sFF[DFF_];
    __shared__ float sRed[H_];
    __shared__ float sRed2[H_];

    const int bn  = blockIdx.x;
    const int b   = bn / N_;
    const int n   = bn % N_;
    const int tid = threadIdx.x;
    const int lane = tid & 31;
    const int wave = tid >> 5;

    const size_t rowBase = (size_t)b * N_ + n;
    const float* heRow = he + rowBase * (K_ * NIN_);

    // ---- stage h_e tile (30x256 f32 -> bf16 in LDS), zero pad rows 30,31 ----
    for (int idx = tid; idx < K_ * NIN_; idx += 256) {
        int k = idx >> 8, c = idx & 255;
        sHe[k * HE_LD + c] = (__bf16)heRow[idx];
    }
    for (int idx = tid; idx < 2 * NIN_; idx += 256) {
        int k = K_ + (idx >> 8), c = idx & 255;
        sHe[k * HE_LD + c] = (__bf16)0.0f;
    }
    if (tid < H_) sHv[tid] = hv[rowBase * H_ + tid];
    if (tid < 32) sMask[tid] = (tid < K_) ? mask_attend[rowBase * K_ + tid] : 0.0f;
    __syncthreads();

    // ---- Q = W_Q @ h_v_row (f32 matvec, threads 0..127) ----
    if (tid < H_) {
        const float* wq = WQ + tid * H_;
        float acc = 0.0f;
#pragma unroll 4
        for (int j = 0; j < H_; ++j) acc = fmaf(wq[j], sHv[j], acc);
        sQ[tid] = acc;
    }

    // ---- K' / V' projections via bf16 WMMA: out(32x128) = he(32x256) @ W^T ----
    {
        const int half = lane >> 4;
        const int l16  = lane & 15;
        for (int t = wave * 4; t < wave * 4 + 4; ++t) {   // 32 tiles over 8 waves
            const int proj = t >> 4;          // 0 = K', 1 = V'
            const int rem  = t & 15;
            const int mt   = rem >> 3;        // 0..1
            const int nt   = rem & 7;         // 0..7
            const __bf16* wrow = (proj ? WVb : WKb) + (size_t)(nt * 16 + l16) * NIN_;
            const __bf16* arow = &sHe[(mt * 16 + l16) * HE_LD];
            v8f acc = {};
#pragma unroll
            for (int kt = 0; kt < 8; ++kt) {
                // A fragment (16x32): K contiguous per half-wave, K = {16h..16h+15}
                v8bf alo = *(const v8bf*)(arow + kt * 32 + 16 * half);
                v8bf ahi = *(const v8bf*)(arow + kt * 32 + 16 * half + 8);
                v16bf a = __builtin_shufflevector(alo, ahi,
                    0, 1, 2, 3, 4, 5, 6, 7, 8, 9, 10, 11, 12, 13, 14, 15);
                // B fragment (32x16): lane holds column, K contiguous per half-wave
                v16bf bf = *(const v16bf*)(wrow + kt * 32 + 16 * half);
                acc = __builtin_amdgcn_wmma_f32_16x16x32_bf16(
                        false, a, false, bf, (short)0, acc, false, false);
            }
            float* dst = proj ? sV : sK;
            const int col = nt * 16 + l16;
#pragma unroll
            for (int r = 0; r < 8; ++r) {
                int row = mt * 16 + r + 8 * half;
                dst[row * KV_LD + col] = acc[r];
            }
        }
    }
    __syncthreads();

    // ---- logits[h][k] = (Q_h . K'_k,h) / sqrt(32), masked ----
    if (tid < NH_ * K_) {
        int h = tid / K_, k = tid % K_;
        const float* qh = &sQ[h * 32];
        const float* kr = &sK[k * KV_LD + h * 32];
        float acc = 0.0f;
#pragma unroll
        for (int j = 0; j < 32; ++j) acc = fmaf(qh[j], kr[j], acc);
        float lg = acc * 0.17677669529663687f;  // 1/sqrt(32)
        sLog[h * 32 + k] = (sMask[k] > 0.0f) ? lg : -3.402823466e38f;
    }
    __syncthreads();

    // ---- softmax over k per head (4 threads) ----
    if (tid < NH_) {
        int h = tid;
        float m = -3.402823466e38f;
        for (int k = 0; k < K_; ++k) m = fmaxf(m, sLog[h * 32 + k]);
        float s = 0.0f;
        for (int k = 0; k < K_; ++k) {
            float ev = __expf(sLog[h * 32 + k] - m);
            sAtt[h * 32 + k] = ev;
            s += ev;
        }
        float inv = 1.0f / s;
        for (int k = 0; k < K_; ++k) sAtt[h * 32 + k] *= inv * sMask[k];
    }
    __syncthreads();

    // ---- h_upd[i] = sum_k att[h][k] * V'[k][i] ----
    if (tid < H_) {
        int h = tid >> 5;
        float acc = 0.0f;
        for (int k = 0; k < K_; ++k) acc = fmaf(sAtt[h * 32 + k], sV[k * KV_LD + tid], acc);
        sHupd[tid] = acc;
    }
    __syncthreads();

    // ---- dh = W_O @ h_upd, residual, stats for LN0 ----
    if (tid < H_) {
        const float* wo = WO + tid * H_;
        float acc = 0.0f;
#pragma unroll 4
        for (int j = 0; j < H_; ++j) acc = fmaf(wo[j], sHupd[j], acc);
        float x = sHv[tid] + acc;
        sH[tid] = x;
        sRed[tid] = x;
        sRed2[tid] = x * x;
    }
    __syncthreads();
    for (int off = 64; off >= 1; off >>= 1) {
        if (tid < off) { sRed[tid] += sRed[tid + off]; sRed2[tid] += sRed2[tid + off]; }
        __syncthreads();
    }
    {
        float mu  = sRed[0] * (1.0f / 128.0f);
        float var = (sRed2[0] - 128.0f * mu * mu) * (1.0f / 127.0f);
        float sig = sqrtf(var + EPS_) + EPS_;
        if (tid < H_) sH[tid] = gain0[tid] * (sH[tid] - mu) / sig + bias0[tid];
    }
    __syncthreads();

    // ---- FFN layer 1: 512 rows, relu ----
    for (int r = tid; r < DFF_; r += 256) {
        const float* wi = Win + (size_t)r * H_;
        float acc = bin[r];
#pragma unroll 4
        for (int j = 0; j < H_; ++j) acc = fmaf(wi[j], sH[j], acc);
        sFF[r] = fmaxf(acc, 0.0f);
    }
    __syncthreads();

    // ---- FFN layer 2 + residual, stats for LN1 ----
    if (tid < H_) {
        const float* wo2 = Wout + (size_t)tid * DFF_;
        float acc = bout[tid];
#pragma unroll 4
        for (int j = 0; j < DFF_; ++j) acc = fmaf(wo2[j], sFF[j], acc);
        float x = sH[tid] + acc;
        sHupd[tid] = x;   // reuse as LN1 input
        sRed[tid] = x;
        sRed2[tid] = x * x;
    }
    __syncthreads();
    for (int off = 64; off >= 1; off >>= 1) {
        if (tid < off) { sRed[tid] += sRed[tid + off]; sRed2[tid] += sRed2[tid + off]; }
        __syncthreads();
    }
    {
        float mu  = sRed[0] * (1.0f / 128.0f);
        float var = (sRed2[0] - 128.0f * mu * mu) * (1.0f / 127.0f);
        float sig = sqrtf(var + EPS_) + EPS_;
        if (tid < H_) {
            float xn = gain1[tid] * (sHupd[tid] - mu) / sig + bias1[tid];
            out[rowBase * H_ + tid] = mask_v[rowBase] * xn;
        }
    }
}

extern "C" void kernel_launch(void* const* d_in, const int* in_sizes, int n_in,
                              void* d_out, int out_size, void* d_ws, size_t ws_size,
                              hipStream_t stream) {
    const float* hv          = (const float*)d_in[0];
    const float* he          = (const float*)d_in[1];
    const float* mask_v      = (const float*)d_in[2];
    const float* mask_attend = (const float*)d_in[3];
    const float* WQ          = (const float*)d_in[4];
    const float* WK          = (const float*)d_in[5];
    const float* WV          = (const float*)d_in[6];
    const float* WO          = (const float*)d_in[7];
    const float* gain0       = (const float*)d_in[8];
    const float* bias0       = (const float*)d_in[9];
    const float* gain1       = (const float*)d_in[10];
    const float* bias1       = (const float*)d_in[11];
    const float* Win         = (const float*)d_in[12];
    const float* bin         = (const float*)d_in[13];
    const float* Wout        = (const float*)d_in[14];
    const float* bout        = (const float*)d_in[15];

    __bf16* wkb = (__bf16*)d_ws;                 // 128*256 bf16 = 64 KB
    __bf16* wvb = wkb + H_ * NIN_;               // + 64 KB  (ws usage: 128 KB)

    cvt_weights<<<(H_ * NIN_ + 255) / 256, 256, 0, stream>>>(WK, WV, wkb, wvb);

    gt_layer<<<B_ * N_, 256, 0, stream>>>(
        hv, he, mask_v, mask_attend, WQ, WO,
        gain0, bias0, gain1, bias1, Win, bin, Wout, bout,
        wkb, wvb, (float*)d_out);
}
